// RSMTMamba_24721831756349
// MI455X (gfx1250) — matrix operand order repaired and checked by reading, not compiled, not run on hardware
//
#include <hip/hip_runtime.h>
#include <math.h>

// ---------------- problem constants ----------------
static constexpr int Bn = 4, Cc = 128, Hh = 48, Ww = 48;
static constexpr int Ll = Hh * Ww;          // 2304
static constexpr int NP = Bn * Ll;          // 9216 pixels
static constexpr int Dd = 256;              // D_INNER
static constexpr int NS = 16;               // D_STATE
static constexpr int RNK = 8;               // DT_RANK
static constexpr int KDIR = 4;              // scan directions
static constexpr int XD = 48;               // padded DT_RANK + 2*D_STATE (40 -> 48)

#define DEVI __device__ __forceinline__

typedef __attribute__((ext_vector_type(16))) _Float16 v16h;
typedef __attribute__((ext_vector_type(8)))  _Float16 v8h;
typedef __attribute__((ext_vector_type(8)))  float    v8f;

// ---------------- helpers ----------------
DEVI int gmap(int k, int l) {
  switch (k) {
    case 0:  return l;
    case 1:  return (l % Hh) * Ww + (l / Hh);
    case 2:  return Ll - 1 - l;
    default: { int j = Ll - 1 - l; return (j % Hh) * Ww + (j / Hh); }
  }
}
DEVI float fast_rcp(float x)  { return __builtin_amdgcn_rcpf(x); }
DEVI float sigm(float x)      { return fast_rcp(1.0f + __expf(-x)); }
DEVI float silu_f(float x)    { return x * sigm(x); }
DEVI float softplus_f(float x){ return (x > 20.0f) ? x : __logf(1.0f + __expf(x)); }
DEVI float gelu_f(float x)    { return 0.5f * x * (1.0f + erff(x * 0.70710678118654752f)); }

DEVI float wave_sum(float v) {
#pragma unroll
  for (int m = 16; m > 0; m >>= 1) v += __shfl_xor(v, m, 32);
  return v;
}

// A-fragment: 16x32 f16, row-major source (lda halves per row).
DEVI v16h load_a_frag(const _Float16* A, int lda, int row0, int k0, int lane) {
  int r = lane & 15, hi = lane >> 4;
  const _Float16* p = A + (size_t)(row0 + r) * lda + k0 + hi * 8;
  v8h lo = *(const v8h*)p;
  v8h hh = *(const v8h*)(p + 16);
  v16h a;
#pragma unroll
  for (int i = 0; i < 8; i++) { a[i] = lo[i]; a[i + 8] = hh[i]; }
  return a;
}
DEVI v16h load_a_frag_row(const _Float16* rowbase, int k0, int hi) {
  const _Float16* p = rowbase + k0 + hi * 8;
  v8h lo = *(const v8h*)p;
  v8h hh = *(const v8h*)(p + 16);
  v16h a;
#pragma unroll
  for (int i = 0; i < 8; i++) { a[i] = lo[i]; a[i + 8] = hh[i]; }
  return a;
}
// B-fragment from native (N_out rows, K cols) row-major weights
DEVI v16h load_b_frag(const _Float16* Bw, int ldb, int n0, int k0, int lane) {
  int n = lane & 15, hi = lane >> 4;
  const _Float16* p = Bw + (size_t)(n0 + n) * ldb + k0 + hi * 8;
  v8h lo = *(const v8h*)p;
  v8h hh = *(const v8h*)(p + 16);
  v16h b;
#pragma unroll
  for (int i = 0; i < 8; i++) { b[i] = lo[i]; b[i + 8] = hh[i]; }
  return b;
}
DEVI v8f wmma_step(v16h a, v16h b, v8f c) {
  return __builtin_amdgcn_wmma_f32_16x16x32_f16(false, a, false, b, (short)0, c,
                                                false, false);
}

// ---------------- weight repack kernels ----------------
__global__ void k_f32tof16(const float* __restrict__ s, _Float16* __restrict__ d, int n) {
  int i = blockIdx.x * blockDim.x + threadIdx.x;
  if (i < n) d[i] = (_Float16)s[i];
}
__global__ void k_wxproj(const float* __restrict__ s, _Float16* __restrict__ d) {
  int i = blockIdx.x * blockDim.x + threadIdx.x;
  if (i >= KDIR * XD * Dd) return;
  int c = i & 255; int r = (i >> 8) % XD; int k = i / (XD * Dd);
  d[i] = (r < RNK + 2 * NS) ? (_Float16)s[((size_t)k * 40 + r) * Dd + c] : (_Float16)0.0f;
}
__global__ void k_wffn(const float* __restrict__ s, _Float16* __restrict__ d) {
  int i = blockIdx.x * blockDim.x + threadIdx.x;
  if (i >= Cc * 1152) return;
  int kk = i % 1152; int co = i / 1152;
  int j = kk >> 7, ci = kk & 127; int ky = j / 3, kx = j % 3;
  d[i] = (_Float16)s[(((size_t)co * Cc + ci) * 3 + ky) * 3 + kx];
}

// ---------------- LN1 (wave per pixel, wave32) ----------------
__global__ void k_ln1(const float* __restrict__ x, const float* __restrict__ g,
                      const float* __restrict__ bb, _Float16* __restrict__ h16) {
  int wid = (blockIdx.x * blockDim.x + threadIdx.x) >> 5;
  int lane = threadIdx.x & 31;
  if (wid >= NP) return;
  int b = wid / Ll, l = wid % Ll, y = l / Ww, xc = l % Ww;
  float v[4]; float s = 0.0f;
#pragma unroll
  for (int j = 0; j < 4; j++) {
    int c = lane + 32 * j;
    v[j] = x[(((size_t)b * Cc + c) * Hh + y) * Ww + xc];
    s += v[j];
  }
  float mean = wave_sum(s) * (1.0f / Cc);
  float q = 0.0f;
#pragma unroll
  for (int j = 0; j < 4; j++) { float dd = v[j] - mean; q += dd * dd; }
  float rs = rsqrtf(wave_sum(q) * (1.0f / Cc) + 1e-5f);
#pragma unroll
  for (int j = 0; j < 4; j++) {
    int c = lane + 32 * j;
    h16[(size_t)wid * Cc + c] = (_Float16)((v[j] - mean) * rs * g[c] + bb[c]);
  }
}

// ---------------- GEMM: in_proj, 16x64 strip per wave (9216 x 512 x 128) ----------------
__global__ void k_gemm_in(const _Float16* __restrict__ A, const _Float16* __restrict__ W,
                          float* __restrict__ xm, float* __restrict__ zb) {
  int lane = threadIdx.x & 31, wave = threadIdx.x >> 5;
  int tile = blockIdx.x * 8 + wave;
  const int MT = NP / 16, NSTR = 512 / 64;
  if (tile >= MT * NSTR) return;
  int m0 = (tile / NSTR) * 16, n0 = (tile % NSTR) * 64;
  v8f acc[4] = {{}, {}, {}, {}};
#pragma unroll
  for (int k0 = 0; k0 < 128; k0 += 32) {
    __builtin_prefetch(W + (size_t)n0 * 128 + k0 + 32, 0, 1);
    v16h a = load_a_frag(A, 128, m0, k0, lane);
#pragma unroll
    for (int t = 0; t < 4; t++)
      acc[t] = wmma_step(a, load_b_frag(W, 128, n0 + 16 * t, k0, lane), acc[t]);
  }
  int mb = m0 + ((lane >> 4) << 3);
  if (n0 < Dd) {            // xm strips (wave-uniform)
#pragma unroll
    for (int t = 0; t < 4; t++) {
      int n = n0 + 16 * t + (lane & 15);
#pragma unroll
      for (int f = 0; f < 8; f++) xm[(size_t)(mb + f) * Dd + n] = acc[t][f];
    }
  } else {                  // z strips -> silu(z)
#pragma unroll
    for (int t = 0; t < 4; t++) {
      int n = n0 - Dd + 16 * t + (lane & 15);
#pragma unroll
      for (int f = 0; f < 8; f++) zb[(size_t)(mb + f) * Dd + n] = silu_f(acc[t][f]);
    }
  }
}

// ---------------- depthwise 3x3 + bias + SiLU ----------------
__global__ void k_dwconv(const float* __restrict__ xm, const float* __restrict__ cw,
                         const float* __restrict__ cb, float* __restrict__ xo32,
                         _Float16* __restrict__ xo16) {
  size_t idx = (size_t)blockIdx.x * blockDim.x + threadIdx.x;
  if (idx >= (size_t)NP * Dd) return;
  int d = (int)(idx & 255);
  size_t p = idx >> 8;
  int b = (int)(p / Ll), l = (int)(p % Ll), y = l / Ww, xc = l % Ww;
  float acc = cb[d];
#pragma unroll
  for (int ky = 0; ky < 3; ky++)
#pragma unroll
    for (int kx = 0; kx < 3; kx++) {
      int yy = y + ky - 1, xx = xc + kx - 1;
      if (yy >= 0 && yy < Hh && xx >= 0 && xx < Ww)
        acc += cw[d * 9 + ky * 3 + kx] * xm[((size_t)b * Ll + yy * Ww + xx) * Dd + d];
    }
  float s = silu_f(acc);
  xo32[idx] = s;
  xo16[idx] = (_Float16)s;
}

// ---------------- GEMM: x_proj per direction, 16x48 strip per wave ----------------
__global__ void k_gemm_xproj(const _Float16* __restrict__ xc16,
                             const _Float16* __restrict__ Wx, float* __restrict__ xdbl) {
  int lane = threadIdx.x & 31, wave = threadIdx.x >> 5;
  int tile = blockIdx.x * 8 + wave;
  const int MT = NP / 16;
  if (tile >= KDIR * MT) return;
  int k = tile / MT;
  int m0 = (tile % MT) * 16;
  int r = lane & 15, hi = lane >> 4;
  int p = m0 + r;                                // tiles never cross batch (L%16==0)
  int b = p / Ll, l = p % Ll;
  const _Float16* arow = xc16 + ((size_t)b * Ll + gmap(k, l)) * Dd;
  const _Float16* Wk = Wx + (size_t)k * XD * Dd;
  v8f acc[3] = {{}, {}, {}};
#pragma unroll
  for (int k0 = 0; k0 < Dd; k0 += 32) {
    v16h a = load_a_frag_row(arow, k0, hi);
#pragma unroll
    for (int t = 0; t < 3; t++)
      acc[t] = wmma_step(a, load_b_frag(Wk, Dd, 16 * t, k0, lane), acc[t]);
  }
  int mb = m0 + ((lane >> 4) << 3);
#pragma unroll
  for (int t = 0; t < 3; t++) {
    int n = 16 * t + (lane & 15);
#pragma unroll
    for (int f = 0; f < 8; f++) {
      int p2 = mb + f; int b2 = p2 / Ll, l2 = p2 % Ll;
      xdbl[(((size_t)b2 * KDIR + k) * Ll + l2) * XD + n] = acc[t][f];
    }
  }
}

// ---------------- selective scan: block per (b,k), thread per d ----------------
__global__ void k_scan(const float* __restrict__ xdbl, const float* __restrict__ xc32,
                       const float* __restrict__ A_log, const float* __restrict__ dt_w,
                       const float* __restrict__ dt_b, const float* __restrict__ Ds,
                       float* __restrict__ ys) {
  __shared__ float sm[32 * XD];
  int bk = blockIdx.x;
  int b = bk / KDIR, k = bk % KDIR;
  int d = threadIdx.x;
  float Av[NS];
#pragma unroll
  for (int n = 0; n < NS; n++) Av[n] = -__expf(A_log[((size_t)k * Dd + d) * NS + n]);
  float wdt[RNK];
#pragma unroll
  for (int j = 0; j < RNK; j++) wdt[j] = dt_w[((size_t)k * Dd + d) * RNK + j];
  float dtb = dt_b[k * Dd + d];
  float dsv = Ds[k * Dd + d];
  float h[NS] = {0.0f};
  const float* xdb = xdbl + ((size_t)b * KDIR + k) * Ll * XD;
  float* yrow = ys + ((size_t)b * KDIR + k) * Ll * Dd;
  for (int l0 = 0; l0 < Ll; l0 += 32) {
    __syncthreads();
    for (int i = threadIdx.x; i < 32 * XD; i += 256) sm[i] = xdb[(size_t)l0 * XD + i];
    __syncthreads();
    for (int li = 0; li < 32; li++) {
      int l = l0 + li;
      const float* row = sm + li * XD;
      float dt = dtb;
#pragma unroll
      for (int j = 0; j < RNK; j++) dt += row[j] * wdt[j];
      dt = softplus_f(dt);
      float xt = xc32[((size_t)b * Ll + gmap(k, l)) * Dd + d];
      float dx = dt * xt;
      float yv = dsv * xt;
#pragma unroll
      for (int n = 0; n < NS; n++) {
        h[n] = h[n] * __expf(dt * Av[n]) + dx * row[8 + n];
        yv += h[n] * row[24 + n];
      }
      yrow[(size_t)l * Dd + d] = yv;
    }
  }
}

// ---------------- merge 4 directions + out-LN * silu(z) -> f16 ----------------
__global__ void k_merge(const float* __restrict__ ys, const float* __restrict__ zb,
                        const float* __restrict__ g, const float* __restrict__ bb,
                        _Float16* __restrict__ a16) {
  int wid = (blockIdx.x * blockDim.x + threadIdx.x) >> 5;
  int lane = threadIdx.x & 31;
  if (wid >= NP) return;
  int b = wid / Ll, l = wid % Ll, y = l / Ww, xc = l % Ww;
  int l1 = xc * Hh + y;
  int l2 = Ll - 1 - l;
  int l3 = Ll - 1 - l1;
  const float* y0 = ys + (((size_t)b * KDIR + 0) * Ll + l ) * Dd;
  const float* y1 = ys + (((size_t)b * KDIR + 1) * Ll + l1) * Dd;
  const float* y2 = ys + (((size_t)b * KDIR + 2) * Ll + l2) * Dd;
  const float* y3 = ys + (((size_t)b * KDIR + 3) * Ll + l3) * Dd;
  float v[8]; float s = 0.0f;
#pragma unroll
  for (int j = 0; j < 8; j++) {
    int d = lane + 32 * j;
    v[j] = y0[d] + y1[d] + y2[d] + y3[d];
    s += v[j];
  }
  float mean = wave_sum(s) * (1.0f / Dd);
  float q = 0.0f;
#pragma unroll
  for (int j = 0; j < 8; j++) { float dd = v[j] - mean; q += dd * dd; }
  float rs = rsqrtf(wave_sum(q) * (1.0f / Dd) + 1e-5f);
#pragma unroll
  for (int j = 0; j < 8; j++) {
    int d = lane + 32 * j;
    float a = (v[j] - mean) * rs * g[d] + bb[d];
    a16[(size_t)wid * Dd + d] = (_Float16)(a * zb[(size_t)wid * Dd + d]);
  }
}

// ---------------- GEMM: out_proj + residual, 16x64 strip (9216 x 128 x 256) ----------------
__global__ void k_gemm_out(const _Float16* __restrict__ A, const _Float16* __restrict__ W,
                           const float* __restrict__ x, _Float16* __restrict__ vss16) {
  int lane = threadIdx.x & 31, wave = threadIdx.x >> 5;
  int tile = blockIdx.x * 8 + wave;
  const int MT = NP / 16, NSTR = Cc / 64;
  if (tile >= MT * NSTR) return;
  int m0 = (tile / NSTR) * 16, n0 = (tile % NSTR) * 64;
  v8f acc[4] = {{}, {}, {}, {}};
#pragma unroll
  for (int k0 = 0; k0 < Dd; k0 += 32) {
    v16h a = load_a_frag(A, Dd, m0, k0, lane);
#pragma unroll
    for (int t = 0; t < 4; t++)
      acc[t] = wmma_step(a, load_b_frag(W, Dd, n0 + 16 * t, k0, lane), acc[t]);
  }
  int mb = m0 + ((lane >> 4) << 3);
#pragma unroll
  for (int t = 0; t < 4; t++) {
    int n = n0 + 16 * t + (lane & 15);
#pragma unroll
    for (int f = 0; f < 8; f++) {
      int m = mb + f;
      int b = m / Ll, l = m % Ll, y = l / Ww, xc = l % Ww;
      float v = acc[t][f] + x[(((size_t)b * Cc + n) * Hh + y) * Ww + xc];
      vss16[(size_t)m * Cc + n] = (_Float16)v;
    }
  }
}

// ---------------- GEMM: ffn 3x3 conv im2col (K=1152) + BN + ReLU6, 16x64 strip ----------------
__global__ void k_gemm_ffn(const _Float16* __restrict__ vss16, const _Float16* __restrict__ Wf,
                           const float* __restrict__ bn_g, const float* __restrict__ bn_b,
                           const float* __restrict__ bn_m, const float* __restrict__ bn_v,
                           _Float16* __restrict__ g16) {
  int lane = threadIdx.x & 31, wave = threadIdx.x >> 5;
  int tile = blockIdx.x * 8 + wave;
  const int MT = NP / 16, NSTR = Cc / 64;
  if (tile >= MT * NSTR) return;
  int m0 = (tile / NSTR) * 16, n0 = (tile % NSTR) * 64;
  int r = lane & 15, hi = lane >> 4;
  int p = m0 + r;
  int b = p / Ll, l = p % Ll, y = l / Ww, xc = l % Ww;
  v8f acc[4] = {{}, {}, {}, {}};
  for (int k0 = 0; k0 < 1152; k0 += 32) {
    v16h a;
#pragma unroll
    for (int gdx = 0; gdx < 2; gdx++) {
      int kk = k0 + hi * 8 + gdx * 16;           // 8-contiguous K, same conv tap
      int j = kk >> 7, ci = kk & 127;
      int ky = j / 3, kx = j % 3;
      int yy = y + ky - 1, xx = xc + kx - 1;
      v8h seg = {};
      if (yy >= 0 && yy < Hh && xx >= 0 && xx < Ww)
        seg = *(const v8h*)(vss16 + ((size_t)b * Ll + yy * Ww + xx) * Cc + ci);
#pragma unroll
      for (int i = 0; i < 8; i++) a[gdx * 8 + i] = seg[i];
    }
    __builtin_prefetch(Wf + (size_t)n0 * 1152 + k0 + 64, 0, 1);
#pragma unroll
    for (int t = 0; t < 4; t++)
      acc[t] = wmma_step(a, load_b_frag(Wf, 1152, n0 + 16 * t, k0, lane), acc[t]);
  }
  int mb = m0 + ((lane >> 4) << 3);
#pragma unroll
  for (int t = 0; t < 4; t++) {
    int n = n0 + 16 * t + (lane & 15);
    float sc = bn_g[n] * rsqrtf(bn_v[n] + 1e-5f);
    float sh = bn_b[n] - bn_m[n] * sc;
#pragma unroll
    for (int f = 0; f < 8; f++) {
      float v = acc[t][f] * sc + sh;
      v = fminf(fmaxf(v, 0.0f), 6.0f);
      g16[(size_t)(mb + f) * Cc + n] = (_Float16)v;
    }
  }
}

// ---------------- GEMM: fc1 (9216 x 64 x 128) + exact GELU, 16x64 strip ----------------
__global__ void k_gemm_fc1(const _Float16* __restrict__ A, const _Float16* __restrict__ W,
                           _Float16* __restrict__ fcbuf) {
  int lane = threadIdx.x & 31, wave = threadIdx.x >> 5;
  int tile = blockIdx.x * 8 + wave;
  const int MT = NP / 16;
  if (tile >= MT) return;
  int m0 = tile * 16;
  v8f acc[4] = {{}, {}, {}, {}};
#pragma unroll
  for (int k0 = 0; k0 < 128; k0 += 32) {
    v16h a = load_a_frag(A, 128, m0, k0, lane);
#pragma unroll
    for (int t = 0; t < 4; t++)
      acc[t] = wmma_step(a, load_b_frag(W, 128, 16 * t, k0, lane), acc[t]);
  }
  int mb = m0 + ((lane >> 4) << 3);
#pragma unroll
  for (int t = 0; t < 4; t++) {
    int n = 16 * t + (lane & 15);
#pragma unroll
    for (int f = 0; f < 8; f++)
      fcbuf[(size_t)(mb + f) * 64 + n] = (_Float16)gelu_f(acc[t][f]);
  }
}

// ---------------- GEMM: fc2 (9216 x 128 x 64) -> NCHW f32 output, 16x64 strip ----------------
__global__ void k_gemm_fc2(const _Float16* __restrict__ A, const _Float16* __restrict__ W,
                           float* __restrict__ out) {
  int lane = threadIdx.x & 31, wave = threadIdx.x >> 5;
  int tile = blockIdx.x * 8 + wave;
  const int MT = NP / 16, NSTR = Cc / 64;
  if (tile >= MT * NSTR) return;
  int m0 = (tile / NSTR) * 16, n0 = (tile % NSTR) * 64;
  v8f acc[4] = {{}, {}, {}, {}};
#pragma unroll
  for (int k0 = 0; k0 < 64; k0 += 32) {
    v16h a = load_a_frag(A, 64, m0, k0, lane);
#pragma unroll
    for (int t = 0; t < 4; t++)
      acc[t] = wmma_step(a, load_b_frag(W, 64, n0 + 16 * t, k0, lane), acc[t]);
  }
  int mb = m0 + ((lane >> 4) << 3);
#pragma unroll
  for (int t = 0; t < 4; t++) {
    int n = n0 + 16 * t + (lane & 15);
#pragma unroll
    for (int f = 0; f < 8; f++) {
      int m = mb + f;
      int b = m / Ll, l = m % Ll, y = l / Ww, xc = l % Ww;
      out[(((size_t)b * Cc + n) * Hh + y) * Ww + xc] = acc[t][f];
    }
  }
}

// ---------------- host launch ----------------
extern "C" void kernel_launch(void* const* d_in, const int* in_sizes, int n_in,
                              void* d_out, int out_size, void* d_ws, size_t ws_size,
                              hipStream_t stream) {
  const float* x         = (const float*)d_in[0];
  const float* ln1_g     = (const float*)d_in[1];
  const float* ln1_b     = (const float*)d_in[2];
  const float* in_proj_w = (const float*)d_in[3];
  const float* conv_w    = (const float*)d_in[4];
  const float* conv_b    = (const float*)d_in[5];
  const float* x_proj_w  = (const float*)d_in[6];
  const float* dt_w      = (const float*)d_in[7];
  const float* dt_b      = (const float*)d_in[8];
  const float* A_log     = (const float*)d_in[9];
  const float* Ds        = (const float*)d_in[10];
  const float* outn_g    = (const float*)d_in[11];
  const float* outn_b    = (const float*)d_in[12];
  const float* out_proj_w= (const float*)d_in[13];
  const float* ffn_conv_w= (const float*)d_in[14];
  const float* bn_g      = (const float*)d_in[15];
  const float* bn_b      = (const float*)d_in[16];
  const float* bn_m      = (const float*)d_in[17];
  const float* bn_v      = (const float*)d_in[18];
  const float* fc1_w     = (const float*)d_in[19];
  const float* fc2_w     = (const float*)d_in[20];
  float* out = (float*)d_out;

  char* base = (char*)d_ws;
  size_t off = 0;
  auto carve = [&](size_t bytes) -> void* {
    off = (off + 255) & ~(size_t)255;
    void* p = base + off;
    off += bytes;
    return p;
  };
  _Float16* h16    = (_Float16*)carve((size_t)NP * Cc * 2);
  _Float16* Win16  = (_Float16*)carve((size_t)512 * 128 * 2);
  _Float16* Wxp16  = (_Float16*)carve((size_t)KDIR * XD * Dd * 2);
  _Float16* Wout16 = (_Float16*)carve((size_t)Cc * Dd * 2);
  _Float16* Wffn16 = (_Float16*)carve((size_t)Cc * 1152 * 2);
  _Float16* Wfc1   = (_Float16*)carve((size_t)64 * 128 * 2);
  _Float16* Wfc2   = (_Float16*)carve((size_t)128 * 64 * 2);
  float*    xmbuf  = (float*)   carve((size_t)NP * Dd * 4);
  float*    zbuf   = (float*)   carve((size_t)NP * Dd * 4);
  float*    xc32   = (float*)   carve((size_t)NP * Dd * 4);
  _Float16* xc16   = (_Float16*)carve((size_t)NP * Dd * 2);
  float*    xdbl   = (float*)   carve((size_t)Bn * KDIR * Ll * XD * 4);
  float*    ysbuf  = (float*)   carve((size_t)Bn * KDIR * Ll * Dd * 4);
  _Float16* a16    = (_Float16*)carve((size_t)NP * Dd * 2);
  _Float16* vss16  = (_Float16*)carve((size_t)NP * Cc * 2);
  _Float16* g16    = (_Float16*)carve((size_t)NP * Cc * 2);
  _Float16* fcbuf  = (_Float16*)carve((size_t)NP * 64 * 2);
  (void)ws_size; (void)in_sizes; (void)n_in; (void)out_size;

  // weight repack
  k_f32tof16<<<(512 * 128 + 255) / 256, 256, 0, stream>>>(in_proj_w, Win16, 512 * 128);
  k_wxproj  <<<(KDIR * XD * Dd + 255) / 256, 256, 0, stream>>>(x_proj_w, Wxp16);
  k_f32tof16<<<(Cc * Dd + 255) / 256, 256, 0, stream>>>(out_proj_w, Wout16, Cc * Dd);
  k_wffn    <<<(Cc * 1152 + 255) / 256, 256, 0, stream>>>(ffn_conv_w, Wffn16);
  k_f32tof16<<<(64 * 128 + 255) / 256, 256, 0, stream>>>(fc1_w, Wfc1, 64 * 128);
  k_f32tof16<<<(128 * 64 + 255) / 256, 256, 0, stream>>>(fc2_w, Wfc2, 128 * 64);

  // pipeline
  k_ln1<<<NP / 8, 256, 0, stream>>>(x, ln1_g, ln1_b, h16);
  { int waves = (NP / 16) * (512 / 64);
    k_gemm_in<<<(waves + 7) / 8, 256, 0, stream>>>(h16, Win16, xmbuf, zbuf); }
  k_dwconv<<<(NP * Dd) / 256, 256, 0, stream>>>(xmbuf, conv_w, conv_b, xc32, xc16);
  { int waves = KDIR * (NP / 16);
    k_gemm_xproj<<<(waves + 7) / 8, 256, 0, stream>>>(xc16, Wxp16, xdbl); }
  k_scan<<<Bn * KDIR, 256, 0, stream>>>(xdbl, xc32, A_log, dt_w, dt_b, Ds, ysbuf);
  k_merge<<<NP / 8, 256, 0, stream>>>(ysbuf, zbuf, outn_g, outn_b, a16);
  { int waves = (NP / 16) * (Cc / 64);
    k_gemm_out<<<(waves + 7) / 8, 256, 0, stream>>>(a16, Wout16, x, vss16); }
  { int waves = (NP / 16) * (Cc / 64);
    k_gemm_ffn<<<(waves + 7) / 8, 256, 0, stream>>>(vss16, Wffn16, bn_g, bn_b, bn_m, bn_v, g16); }
  { int waves = (NP / 16);
    k_gemm_fc1<<<(waves + 7) / 8, 256, 0, stream>>>(g16, Wfc1, fcbuf); }
  { int waves = (NP / 16) * (Cc / 64);
    k_gemm_fc2<<<(waves + 7) / 8, 256, 0, stream>>>(fcbuf, Wfc2, out); }
}
